// SparseSelfAttention_79156247265922
// MI455X (gfx1250) — matrix operand order repaired and checked by reading, not compiled, not run on hardware
//
#include <hip/hip_runtime.h>

#define B_ 4
#define S_ 2048
#define D_ 2048
#define E_ 8
#define HD_ 256
#define KCAP 1024
#define NEB 32
#define QKVN 768
#define SCALE_ 0.0625f

typedef __attribute__((ext_vector_type(16))) __bf16 bf16x16;
typedef __attribute__((ext_vector_type(8)))  __bf16 bf16x8;
typedef __attribute__((ext_vector_type(8)))  float  f32x8;

union ABFrag { bf16x16 v; bf16x8 h[2]; };

static __device__ inline bf16x16 load_a_frag(const __bf16* row, int hx) {
  ABFrag f;
  f.h[0] = *(const bf16x8*)(row + 8 * hx);
  f.h[1] = *(const bf16x8*)(row + 16 + 8 * hx);
  return f.v;
}
static __device__ inline bf16x16 load_b_frag(const __bf16* row, int hx) {
  ABFrag f;
  f.h[0] = *(const bf16x8*)(row + 8 * hx);
  f.h[1] = *(const bf16x8*)(row + 16 + 8 * hx);
  return f.v;
}
static __device__ inline f32x8 wmma_bf16(bf16x16 a, bf16x16 b, f32x8 c) {
  return __builtin_amdgcn_wmma_f32_16x16x32_bf16(false, a, false, b, (short)0, c,
                                                 false, false);
}
static __device__ inline unsigned pack_bf2(float a, float b) {
  __bf16 x = (__bf16)a, y = (__bf16)b;
  unsigned short ux = __builtin_bit_cast(unsigned short, x);
  unsigned short uy = __builtin_bit_cast(unsigned short, y);
  return (unsigned)ux | ((unsigned)uy << 16);
}
static __device__ inline f32x8 zero8() {
  f32x8 z = {0.f, 0.f, 0.f, 0.f, 0.f, 0.f, 0.f, 0.f};
  return z;
}

// -------------------- utility kernels --------------------

__global__ void k_zero_out(float4* out, int n4) {
  int i = blockIdx.x * blockDim.x + threadIdx.x;
  if (i < n4) out[i] = make_float4(0.f, 0.f, 0.f, 0.f);
}

__global__ void k_cvt_x(const float4* X, unsigned short* Xbf, int n4) {
  int i = blockIdx.x * blockDim.x + threadIdx.x;
  if (i < n4) {
    float4 v = X[i];
    unsigned* o = (unsigned*)(Xbf + (size_t)i * 4);
    o[0] = pack_bf2(v.x, v.y);
    o[1] = pack_bf2(v.z, v.w);
  }
}

__global__ void k_prep_wg(const float* w_gate, unsigned short* wgT, int* counts,
                          int* mlen) {
  int i = blockIdx.x * 256 + threadIdx.x;  // 16*2048
  int n = i / D_, k = i % D_;
  float v = (n < E_) ? w_gate[(size_t)k * E_ + n] : 0.f;
  __bf16 bv = (__bf16)v;
  wgT[(size_t)n * D_ + k] = __builtin_bit_cast(unsigned short, bv);
  if (i < NEB) counts[i] = 0;
  if (i == NEB) *mlen = 0;
}

__global__ void k_prep_wqkvT(const float* W_qkv, unsigned short* WT) {
  int idx = blockIdx.x * 256 + threadIdx.x;  // 8*768*2048
  int e = idx / (QKVN * D_);
  int r = idx % (QKVN * D_);
  int n = r / D_, k = r % D_;
  float v = W_qkv[((size_t)e * D_ + k) * QKVN + n];
  __bf16 bv = (__bf16)v;
  WT[idx] = __builtin_bit_cast(unsigned short, bv);
}

__global__ void k_prep_wffT(const float* W_ff, unsigned short* WT) {
  int idx = blockIdx.x * 256 + threadIdx.x;  // 8*2048*256
  int e = idx / (D_ * HD_);
  int r = idx % (D_ * HD_);
  int n = r / HD_, k = r % HD_;
  float v = W_ff[((size_t)e * HD_ + k) * D_ + n];
  __bf16 bv = (__bf16)v;
  WT[idx] = __builtin_bit_cast(unsigned short, bv);
}

// -------------------- routing --------------------

__global__ void __launch_bounds__(256) k_route(const unsigned short* Xbf_u,
                                               const unsigned short* wgT_u,
                                               const float* b_gate,
                                               float* maskedG, int* counts) {
  __shared__ float lds[8 * 16 * 16];
  const __bf16* Xbf = (const __bf16*)Xbf_u;
  const __bf16* wgT = (const __bf16*)wgT_u;
  int tid = threadIdx.x, wave = tid >> 5, lane = tid & 31;
  int hx = lane >> 4, l16 = lane & 15;
  int t0 = (blockIdx.x * 8 + wave) * 16;
  const __bf16* arow = Xbf + (size_t)(t0 + l16) * D_;
  const __bf16* brow = wgT + (size_t)l16 * D_;
  f32x8 c = zero8();
#pragma unroll 4
  for (int k0 = 0; k0 < D_; k0 += 32) {
    bf16x16 a = load_a_frag(arow + k0, hx);
    bf16x16 b = load_b_frag(brow + k0, hx);
    c = wmma_bf16(a, b, c);
  }
#pragma unroll
  for (int r = 0; r < 8; r++)
    lds[wave * 256 + (r + 8 * hx) * 16 + l16] = c[r];
  __syncthreads();
  if (tid < 128) {
    int w = tid >> 4, row = tid & 15;
    int t = (blockIdx.x * 8 + w) * 16 + row;
    float g[E_];
    float mx = -1e30f;
#pragma unroll
    for (int e = 0; e < E_; e++) {
      g[e] = lds[w * 256 + row * 16 + e] + b_gate[e];
      mx = fmaxf(mx, g[e]);
    }
    float sum = 0.f;
#pragma unroll
    for (int e = 0; e < E_; e++) { g[e] = __expf(g[e] - mx); sum += g[e]; }
    float inv = 1.f / sum;
    int i1 = 0;
#pragma unroll
    for (int e = 1; e < E_; e++) if (g[e] > g[i1]) i1 = e;
    int i2 = (i1 == 0) ? 1 : 0;
#pragma unroll
    for (int e = 0; e < E_; e++)
      if (e != i1 && e != i2 && g[e] > g[i2]) i2 = e;
    int b = t / S_;
#pragma unroll
    for (int e = 0; e < E_; e++)
      maskedG[(size_t)t * E_ + e] = (e == i1 || e == i2) ? g[e] * inv : 0.f;
    atomicAdd(&counts[i1 * B_ + b], 1);
    atomicAdd(&counts[i2 * B_ + b], 1);
  }
}

__global__ void k_route_max(const int* counts, int* mlen) {
  if (threadIdx.x == 0) {
    int m = 1;
    for (int i = 0; i < NEB; i++) m = max(m, counts[i]);
    *mlen = min(m, KCAP);
  }
}

__global__ void __launch_bounds__(256) k_compact(const float* maskedG,
                                                 const int* counts,
                                                 const int* mlenp,
                                                 const float* maskin, int* seq,
                                                 float* maskg) {
  __shared__ int wtot[2][8];
  int eb = blockIdx.x;
  int e = eb >> 2, b = eb & 3;
  int mlen = *mlenp;
  int cnt = counts[eb];
  int pad = mlen - cnt; if (pad < 0) pad = 0;
  int tid = threadIdx.x, wave = tid >> 5, lane = tid & 31;
  int run_pos = 0, run_zero = 0;
  for (int c0 = 0; c0 < S_; c0 += 256) {
    int s = c0 + tid;
    bool pos = maskedG[((size_t)b * S_ + s) * E_ + e] > 0.f;
    unsigned bp = (unsigned)__ballot(pos);
    int myrank_pos = __popc(bp & ((lane == 0) ? 0u : ((1u << lane) - 1u)));
    int wp = __popc(bp);
    int myrank_zero = lane - myrank_pos;
    int wz = 32 - wp;
    if (lane == 0) { wtot[0][wave] = wp; wtot[1][wave] = wz; }
    __syncthreads();
    int off_p = 0, off_z = 0;
    for (int w = 0; w < 8; w++)
      if (w < wave) { off_p += wtot[0][w]; off_z += wtot[1][w]; }
    int pos_before = run_pos + off_p + myrank_pos;
    int zero_before = run_zero + off_z + myrank_zero;
    int outp = -1;
    if (pos) outp = pos_before + min(zero_before, pad);
    else if (zero_before < pad) outp = pos_before + zero_before;
    if (outp >= 0 && outp < mlen) {
      seq[eb * KCAP + outp] = s;
      maskg[eb * KCAP + outp] = maskin[(size_t)b * S_ + s];
    }
    int tot_p = 0, tot_z = 0;
    for (int w = 0; w < 8; w++) { tot_p += wtot[0][w]; tot_z += wtot[1][w]; }
    run_pos += tot_p; run_zero += tot_z;
    __syncthreads();
  }
}

// -------------------- QKV projection + RoPE --------------------

__global__ void __launch_bounds__(256) k_qkv(const unsigned short* Xbf_u,
                                             const unsigned short* WT_u,
                                             const float* b_qkv, const int* seq,
                                             const int* mlenp,
                                             unsigned short* qk_u,
                                             unsigned short* vT_u) {
  const __bf16* Xbf = (const __bf16*)Xbf_u;
  const __bf16* WT = (const __bf16*)WT_u;
  __bf16* qk = (__bf16*)qk_u;
  __bf16* vT = (__bf16*)vT_u;
  int mlen = *mlenp;
  int lim = (mlen + 31) & ~31; if (lim > KCAP) lim = KCAP;
  int m0 = blockIdx.y * 16;
  if (m0 >= lim) return;
  int eb = blockIdx.z;
  int e = eb >> 2, b = eb & 3;
  int tid = threadIdx.x, wave = tid >> 5, lane = tid & 31;
  int hx = lane >> 4, l16 = lane & 15;
  int n = blockIdx.x * 128 + wave * 16 + l16;
  int m = m0 + l16;
  int sid = (m < mlen) ? seq[eb * KCAP + m] : seq[eb * KCAP];
  const __bf16* arow = Xbf + ((size_t)b * S_ + sid) * D_;
  const __bf16* brow = WT + ((size_t)e * QKVN + n) * (size_t)D_;
  f32x8 c = zero8();
#pragma unroll 4
  for (int k0 = 0; k0 < D_; k0 += 32) {
    __builtin_prefetch(arow + k0 + 256, 0, 1);
    bf16x16 a = load_a_frag(arow + k0, hx);
    bf16x16 bm = load_b_frag(brow + k0, hx);
    c = wmma_bf16(a, bm, c);
  }
  float bias = b_qkv[e * QKVN + n];
  bool is_pe = (n >= 128 && n < 256) || (n >= 384 && n < 512);
  int base = (n < 256) ? 128 : 384;
  int pidx = ((n & ~1) - base) >> 1;
  // freq = 10000^{-(e*64+i)/512}
  float fr = __expf(-(float)(e * 64 + pidx) * 0.017988946f);
  bool even = ((n & 1) == 0);
#pragma unroll
  for (int r = 0; r < 8; r++) {
    int mm = m0 + r + 8 * hx;
    float val = c[r] + bias;
    if (is_pe) {
      float other = __shfl_xor(val, 1, 32);
      float sn, cn;
      __sincosf((float)mm * fr, &sn, &cn);
      float av = even ? val : other;
      float bv = even ? other : val;
      val = even ? (av * cn - bv * sn) : (av * sn + bv * cn);
    }
    if (mm >= mlen) val = 0.f;  // zero-pad tail rows up to lim
    if (n < 512)
      qk[((size_t)eb * KCAP + mm) * 512 + n] = (__bf16)val;
    else
      vT[((size_t)eb * 256 + (n - 512)) * KCAP + mm] = (__bf16)val;
  }
}

// -------------------- attention (flash-style, S^T tiles) --------------------

__global__ void __launch_bounds__(256, 1) k_attn(const unsigned short* qk_u,
                                                 const unsigned short* vT_u,
                                                 const float* maskg,
                                                 const int* mlenp,
                                                 unsigned short* ctx_u) {
  __shared__ unsigned short Plds[8 * 16 * 32];
  __shared__ float m_arr[8][16];
  __shared__ float l_arr[8][16];
  __shared__ float ctx_lds[16][256];
  const __bf16* qkb = (const __bf16*)qk_u;
  const __bf16* vTb = (const __bf16*)vT_u;
  __bf16* ctx = (__bf16*)ctx_u;
  int mlen = *mlenp;
  int q0 = blockIdx.x * 16;
  if (q0 >= mlen) return;
  int eb = blockIdx.y;
  int tid = threadIdx.x, wave = tid >> 5, lane = tid & 31;
  int hx = lane >> 4, l16 = lane & 15;
  for (int i = tid; i < 16 * 256; i += 256) ((float*)ctx_lds)[i] = 0.f;
  __syncthreads();

  const __bf16* qbase = qkb + (size_t)eb * KCAP * 512;
  const __bf16* kbase = qbase + 256;
  const __bf16* vbase = vTb + (size_t)eb * 256 * KCAP;
  const float* mg = maskg + eb * KCAP;
  const __bf16* qrow = qbase + (size_t)(q0 + l16) * 512;

  float m_run = -1e30f, l_run = 0.f;
  f32x8 acc[16];
#pragma unroll
  for (int ch = 0; ch < 16; ch++) acc[ch] = zero8();

  for (int kb = wave * 32; kb < mlen; kb += 256) {
    const __bf16* krow0 = kbase + (size_t)(kb + l16) * 512;
    const __bf16* krow1 = kbase + (size_t)(kb + 16 + l16) * 512;
    f32x8 c0 = zero8(), c1 = zero8();
#pragma unroll
    for (int d0 = 0; d0 < HD_; d0 += 32) {
      bf16x16 bq = load_b_frag(qrow + d0, hx);
      bf16x16 a0 = load_a_frag(krow0 + d0, hx);
      c0 = wmma_bf16(a0, bq, c0);
      bf16x16 a1 = load_a_frag(krow1 + d0, hx);
      c1 = wmma_bf16(a1, bq, c1);
    }
    // scores (S^T layout: per-lane query = l16, keys across VGPRs/halves)
    float s0[8], s1[8];
    float tmax = -1e30f;
#pragma unroll
    for (int r = 0; r < 8; r++) {
      int key0 = kb + r + 8 * hx;
      int key1 = key0 + 16;
      float ma0 = (key0 < mlen) ? -1.0e6f * (1.f - mg[key0]) : -1.0e6f;
      float ma1 = (key1 < mlen) ? -1.0e6f * (1.f - mg[key1]) : -1.0e6f;
      s0[r] = c0[r] * SCALE_ + ma0;
      s1[r] = c1[r] * SCALE_ + ma1;
      tmax = fmaxf(tmax, fmaxf(s0[r], s1[r]));
    }
    tmax = fmaxf(tmax, __shfl_xor(tmax, 16, 32));
    float m_new = fmaxf(m_run, tmax);
    float corr = __expf(m_run - m_new);
    l_run *= corr;
    // broadcast per-query correction to ctx-row layout (rows r + 8*hx)
    float cb[8];
#pragma unroll
    for (int r = 0; r < 8; r++) cb[r] = __shfl(corr, r + 8 * hx, 32);
#pragma unroll
    for (int ch = 0; ch < 16; ch++)
#pragma unroll
      for (int r = 0; r < 8; r++) acc[ch][r] *= cb[r];

    float psum = 0.f;
    float p0[8], p1[8];
#pragma unroll
    for (int r = 0; r < 8; r++) {
      p0[r] = __expf(s0[r] - m_new);
      p1[r] = __expf(s1[r] - m_new);
      psum += p0[r] + p1[r];
    }
    psum += __shfl_xor(psum, 16, 32);
    l_run += psum;
    m_run = m_new;

    // stash P (bf16) into this wave's LDS tile: row=query(l16), 32 keys
    unsigned* pb = (unsigned*)&Plds[wave * 512 + l16 * 32];
#pragma unroll
    for (int j = 0; j < 4; j++) {
      pb[4 * hx + j] = pack_bf2(p0[2 * j], p0[2 * j + 1]);
      pb[8 + 4 * hx + j] = pack_bf2(p1[2 * j], p1[2 * j + 1]);
    }
    bf16x16 pa = load_a_frag((const __bf16*)&Plds[wave * 512 + l16 * 32], hx);
    // ctx += P @ V   (B-frag from V^T[d][key], contiguous keys)
#pragma unroll
    for (int ch = 0; ch < 16; ch++) {
      const __bf16* vrow = vbase + (size_t)(ch * 16 + l16) * KCAP + kb;
      bf16x16 vb = load_b_frag(vrow, hx);
      acc[ch] = wmma_bf16(pa, vb, acc[ch]);
    }
  }

  if (lane < 16) { m_arr[wave][lane] = m_run; l_arr[wave][lane] = l_run; }
  __syncthreads();

#pragma unroll
  for (int r = 0; r < 8; r++) {
    int q = r + 8 * hx;
    float Mq = -1e30f;
#pragma unroll
    for (int w = 0; w < 8; w++) Mq = fmaxf(Mq, m_arr[w][q]);
    float Lq = 0.f;
#pragma unroll
    for (int w = 0; w < 8; w++) Lq += l_arr[w][q] * __expf(m_arr[w][q] - Mq);
    float f = __expf(m_arr[wave][q] - Mq) / fmaxf(Lq, 1e-30f);
#pragma unroll
    for (int ch = 0; ch < 16; ch++)
      atomicAdd(&ctx_lds[q][ch * 16 + l16], acc[ch][r] * f);
  }
  __syncthreads();

  for (int i = tid; i < 16 * 256; i += 256) {
    int qq = i >> 8, d = i & 255;
    if (q0 + qq < mlen)
      ctx[((size_t)eb * KCAP + q0 + qq) * 256 + d] = (__bf16)ctx_lds[qq][d];
  }
}

// -------------------- output projection + scatter-add --------------------

__global__ void __launch_bounds__(256) k_out(const unsigned short* ctx_u,
                                             const unsigned short* WT_u,
                                             const float* b_ff, const int* seq,
                                             const int* mlenp, float* out) {
  const __bf16* ctx = (const __bf16*)ctx_u;
  const __bf16* WT = (const __bf16*)WT_u;
  int mlen = *mlenp;
  int m0 = blockIdx.y * 16;
  if (m0 >= mlen) return;
  int eb = blockIdx.z;
  int e = eb >> 2, b = eb & 3;
  int tid = threadIdx.x, wave = tid >> 5, lane = tid & 31;
  int hx = lane >> 4, l16 = lane & 15;
  int n = blockIdx.x * 128 + wave * 16 + l16;
  const __bf16* arow = ctx + ((size_t)eb * KCAP + m0 + l16) * 256;
  const __bf16* brow = WT + ((size_t)e * D_ + n) * 256;
  f32x8 c = zero8();
#pragma unroll
  for (int k0 = 0; k0 < HD_; k0 += 32) {
    bf16x16 a = load_a_frag(arow + k0, hx);
    bf16x16 bm = load_b_frag(brow + k0, hx);
    c = wmma_bf16(a, bm, c);
  }
  float bias = b_ff[n];
#pragma unroll
  for (int r = 0; r < 8; r++) {
    int mm = m0 + r + 8 * hx;
    if (mm < mlen) {
      int sid = seq[eb * KCAP + mm];
      atomicAdd(&out[((size_t)b * S_ + sid) * D_ + n], c[r] + bias);
    }
  }
}

// -------------------- host launcher --------------------

extern "C" void kernel_launch(void* const* d_in, const int* in_sizes, int n_in,
                              void* d_out, int out_size, void* d_ws,
                              size_t ws_size, hipStream_t stream) {
  const float* X = (const float*)d_in[0];
  const float* mask = (const float*)d_in[1];
  const float* w_gate = (const float*)d_in[2];
  const float* b_gate = (const float*)d_in[3];
  const float* W_qkv = (const float*)d_in[4];
  const float* b_qkv = (const float*)d_in[5];
  const float* W_ff = (const float*)d_in[6];
  const float* b_ff = (const float*)d_in[7];
  float* out = (float*)d_out;
  char* ws = (char*)d_ws;

  constexpr size_t OFF_MASKED = 0;                                     // 256 KB
  constexpr size_t OFF_COUNTS = OFF_MASKED + (size_t)B_ * S_ * E_ * 4;
  constexpr size_t OFF_MLEN = OFF_COUNTS + 256;
  constexpr size_t OFF_SEQ = OFF_MLEN + 256;
  constexpr size_t OFF_MASKG = OFF_SEQ + (size_t)NEB * KCAP * 4;
  constexpr size_t OFF_XBF = OFF_MASKG + (size_t)NEB * KCAP * 4;
  constexpr size_t OFF_WGT = OFF_XBF + (size_t)B_ * S_ * D_ * 2;
  constexpr size_t OFF_WQKVT = OFF_WGT + (size_t)16 * D_ * 2;
  constexpr size_t OFF_WFFT = OFF_WQKVT + (size_t)E_ * QKVN * D_ * 2;
  constexpr size_t OFF_QK = OFF_WFFT + (size_t)E_ * D_ * HD_ * 2;
  constexpr size_t OFF_VT = OFF_QK + (size_t)NEB * KCAP * 512 * 2;
  constexpr size_t OFF_CTX = OFF_VT + (size_t)NEB * 256 * KCAP * 2;

  unsigned short* xbf = (unsigned short*)(ws + OFF_XBF);
  unsigned short* wgT = (unsigned short*)(ws + OFF_WGT);
  unsigned short* wqkvT = (unsigned short*)(ws + OFF_WQKVT);
  unsigned short* wffT = (unsigned short*)(ws + OFF_WFFT);
  unsigned short* qk = (unsigned short*)(ws + OFF_QK);
  unsigned short* vT = (unsigned short*)(ws + OFF_VT);
  unsigned short* ctx = (unsigned short*)(ws + OFF_CTX);
  float* maskedG = (float*)(ws + OFF_MASKED);
  float* maskg = (float*)(ws + OFF_MASKG);
  int* counts = (int*)(ws + OFF_COUNTS);
  int* mlen = (int*)(ws + OFF_MLEN);
  int* seq = (int*)(ws + OFF_SEQ);

  const int n4 = B_ * S_ * D_ / 4;  // 4,194,304

  k_zero_out<<<dim3(n4 / 256), 256, 0, stream>>>((float4*)out, n4);
  k_cvt_x<<<dim3(n4 / 256), 256, 0, stream>>>((const float4*)X, xbf, n4);
  k_prep_wg<<<dim3(128), 256, 0, stream>>>(w_gate, wgT, counts, mlen);
  k_prep_wqkvT<<<dim3(E_ * QKVN * D_ / 256), 256, 0, stream>>>(W_qkv, wqkvT);
  k_prep_wffT<<<dim3(E_ * D_ * HD_ / 256), 256, 0, stream>>>(W_ff, wffT);
  k_route<<<dim3(B_ * S_ / 128), 256, 0, stream>>>(xbf, wgT, b_gate, maskedG,
                                                   counts);
  k_route_max<<<dim3(1), 32, 0, stream>>>(counts, mlen);
  k_compact<<<dim3(NEB), 256, 0, stream>>>(maskedG, counts, mlen, mask, seq,
                                           maskg);
  k_qkv<<<dim3(QKVN / 128, KCAP / 16, NEB), 256, 0, stream>>>(
      xbf, wqkvT, b_qkv, seq, mlen, qk, vT);
  k_attn<<<dim3(KCAP / 16, NEB), 256, 0, stream>>>(qk, vT, maskg, mlen, ctx);
  k_out<<<dim3(D_ / 128, KCAP / 16, NEB), 256, 0, stream>>>(ctx, wffT, b_ff,
                                                            seq, mlen, out);
  (void)in_sizes; (void)n_in; (void)out_size; (void)ws_size;
}
